// DGCNNWithKNN_78314433675372
// MI455X (gfx1250) — matrix operand order repaired and checked by reading, not compiled, not run on hardware
//
#include <hip/hip_runtime.h>
#include <hip/hip_bf16.h>

#define DEV_INLINE __device__ __forceinline__

typedef __attribute__((ext_vector_type(16))) _Float16 v16h;
typedef __attribute__((ext_vector_type(8)))  _Float16 h8;
typedef __attribute__((ext_vector_type(8)))  float    v8f;

static constexpr int NPTS  = 16384;
static constexpr int KNN   = 20;
static constexpr int NEDGE = NPTS * KNN;   // 327680, /16 = 20480 tiles

// ---------------- WMMA fragment helpers (CDNA5 16x16x32 f16 layouts) ----------------

// A (16x32 f16, row-major source): lane<16 holds row M=lane, K = {kb..kb+7, kb+16..kb+23}
//                                  lane>=16 holds row M=lane-16, K shifted by +8.
DEV_INLINE v16h load_a_frag(const _Float16* tile, int ld, int kb) {
  const int lane = threadIdx.x & 31;
  const int hl = lane & 15, hi = lane >> 4;
  const _Float16* p = tile + hl * ld + kb + hi * 8;
  h8 lo = *(const h8*)(p);
  h8 hh = *(const h8*)(p + 16);
  v16h f;
#pragma unroll
  for (int t = 0; t < 8; ++t) { f[t] = lo[t]; f[t + 8] = hh[t]; }
  return f;
}

// B (32x16 f16) built from a transposed operand Wt stored row-major [M x Kpad]:
// B[k][n] = Wt[n0+n][k].  lane n (<16): K = {kb..kb+7, kb+16..kb+23}; lanes 16-31: +8.
DEV_INLINE v16h load_bt_frag(const _Float16* wt, int ld, int kb) {
  const int lane = threadIdx.x & 31;
  const int hl = lane & 15, hi = lane >> 4;
  const _Float16* p = wt + hl * ld + kb + hi * 8;
  h8 lo = *(const h8*)(p);
  h8 hh = *(const h8*)(p + 16);
  v16h f;
#pragma unroll
  for (int t = 0; t < 8; ++t) { f[t] = lo[t]; f[t + 8] = hh[t]; }
  return f;
}

DEV_INLINE v8f wmma16x16x32(v16h a, v16h b, v8f c) {
  return __builtin_amdgcn_wmma_f32_16x16x32_f16(false, a, false, b, (short)0, c, false, false);
}

// float atomic max via signed/unsigned integer atomics (works for all finite floats, -inf init)
DEV_INLINE void atomicMaxF(float* addr, float val) {
  if (val >= 0.0f) atomicMax((int*)addr, __float_as_int(val));
  else             atomicMin((unsigned int*)addr, (unsigned int)__float_as_int(val));
}

// ---------------- kernels ----------------

// squared norms + f16 copy padded to Dpad columns
__global__ void prep_kernel(const float* __restrict__ x, int n, int D, int Dpad,
                            float* __restrict__ sq, _Float16* __restrict__ xh) {
  int i = blockIdx.x * blockDim.x + threadIdx.x;
  if (i >= n) return;
  float acc = 0.f;
  for (int c = 0; c < Dpad; ++c) {
    float v = (c < D) ? x[i * D + c] : 0.f;
    acc += v * v;
    xh[i * Dpad + c] = (_Float16)v;
  }
  sq[i] = acc;
}

// distance strip: d[i][j] = sq_i + sq_j - 2 * <x_i, x_j> via WMMA Gram tiles.
// 32x64 block per wave (2 row-tiles x 4 col-tiles), explicit double buffering:
// next k-step's 6 fragments are loaded before the current k-step's 8 WMMAs execute.
__global__ void dist_kernel(const _Float16* __restrict__ xh, const float* __restrict__ sq,
                            int n, int Dpad, int row_base, float* __restrict__ dist) {
  const int wid  = threadIdx.x >> 5;
  const int tile = blockIdx.x * 4 + wid;
  const int cgroups = n >> 6;                       // 64-column groups
  const int rt = tile / cgroups, cg = tile - rt * cgroups;
  const int row0 = row_base + rt * 32;
  const _Float16* at0 = xh + (size_t)row0 * Dpad;
  const _Float16* at1 = at0 + (size_t)16 * Dpad;
  const _Float16* bt0 = xh + (size_t)(cg * 64) * Dpad;  // Gram: B = rows of X (transposed use)
  const _Float16* bt1 = bt0 + (size_t)16 * Dpad;
  const _Float16* bt2 = bt0 + (size_t)32 * Dpad;
  const _Float16* bt3 = bt0 + (size_t)48 * Dpad;
  __builtin_prefetch(bt0, 0, 0);                    // global_prefetch_b8
  v8f acc[2][4];
  {
    v8f z = {};
#pragma unroll
    for (int rb = 0; rb < 2; ++rb)
#pragma unroll
      for (int cb = 0; cb < 4; ++cb) acc[rb][cb] = z;
  }
  v16h ca0 = load_a_frag(at0, Dpad, 0);
  v16h ca1 = load_a_frag(at1, Dpad, 0);
  v16h cb0 = load_bt_frag(bt0, Dpad, 0);
  v16h cb1 = load_bt_frag(bt1, Dpad, 0);
  v16h cb2 = load_bt_frag(bt2, Dpad, 0);
  v16h cb3 = load_bt_frag(bt3, Dpad, 0);
  for (int kb = 0; kb < Dpad; kb += 32) {
    const int kn = (kb + 32 < Dpad) ? kb + 32 : kb;   // last iter: harmless reload
    v16h na0 = load_a_frag(at0, Dpad, kn);
    v16h na1 = load_a_frag(at1, Dpad, kn);
    v16h nb0 = load_bt_frag(bt0, Dpad, kn);
    v16h nb1 = load_bt_frag(bt1, Dpad, kn);
    v16h nb2 = load_bt_frag(bt2, Dpad, kn);
    v16h nb3 = load_bt_frag(bt3, Dpad, kn);
    acc[0][0] = wmma16x16x32(ca0, cb0, acc[0][0]);
    acc[1][0] = wmma16x16x32(ca1, cb0, acc[1][0]);
    acc[0][1] = wmma16x16x32(ca0, cb1, acc[0][1]);
    acc[1][1] = wmma16x16x32(ca1, cb1, acc[1][1]);
    acc[0][2] = wmma16x16x32(ca0, cb2, acc[0][2]);
    acc[1][2] = wmma16x16x32(ca1, cb2, acc[1][2]);
    acc[0][3] = wmma16x16x32(ca0, cb3, acc[0][3]);
    acc[1][3] = wmma16x16x32(ca1, cb3, acc[1][3]);
    ca0 = na0; ca1 = na1;
    cb0 = nb0; cb1 = nb1; cb2 = nb2; cb3 = nb3;
  }
  const int lane = threadIdx.x & 31;
  const int hl = lane & 15, hi = lane >> 4;
#pragma unroll
  for (int cb = 0; cb < 4; ++cb) {
    const int col = cg * 64 + cb * 16 + hl;
    const float sqj = sq[col];
#pragma unroll
    for (int rb = 0; rb < 2; ++rb) {
#pragma unroll
      for (int r = 0; r < 8; ++r) {
        int row = row0 + rb * 16 + hi * 8 + r;
        float d = sq[row] + sqj - 2.0f * acc[rb][cb][r];
        d = fmaxf(d, 0.0f);             // keep float bit-order == numeric order for top-k packing
        if (row == col) d = __builtin_inff();
        dist[(size_t)(row - row_base) * n + col] = d;
      }
    }
  }
}

// one wave per row: per-lane sorted top-20 insertion lists, then 20-round k-way merge
__global__ void topk_kernel(const float* __restrict__ dist, int n, int row_base,
                            int rows, int* __restrict__ idx_out) {
  __shared__ unsigned long long cand[4][32 * KNN];
  const int wid  = threadIdx.x >> 5;
  const int lane = threadIdx.x & 31;
  const int rl = blockIdx.x * 4 + wid;
  if (rl >= rows) return;
  const int row = row_base + rl;
  float dk[KNN]; int ik[KNN];
#pragma unroll
  for (int t = 0; t < KNN; ++t) { dk[t] = __builtin_inff(); ik[t] = -1; }
  const float* drow = dist + (size_t)rl * n;
  for (int j = lane; j < n; j += 32) {
    float d = drow[j];
    if (d < dk[KNN - 1]) {
      float cv = d; int ci = j;
#pragma unroll
      for (int t = 0; t < KNN; ++t) {
        if (cv < dk[t]) {
          float tf = dk[t]; dk[t] = cv; cv = tf;
          int ti = ik[t]; ik[t] = ci; ci = ti;
        }
      }
    }
  }
  unsigned long long* my = &cand[wid][lane * KNN];
#pragma unroll
  for (int t = 0; t < KNN; ++t)
    my[t] = ((unsigned long long)__float_as_uint(dk[t]) << 32) | (unsigned int)ik[t];
  int ptr = 0;
  for (int r = 0; r < KNN; ++r) {
    unsigned long long c = (ptr < KNN) ? my[ptr] : ~0ull;
    unsigned long long m = c;
#pragma unroll
    for (int off = 16; off > 0; off >>= 1) {
      unsigned long long o = __shfl_xor(m, off, 32);
      m = (o < m) ? o : m;
    }
    if (c == m) ptr++;                  // unique winner (index uniquifies keys)
    if (lane == 0) idx_out[row * KNN + r] = (int)(unsigned int)(m & 0xffffffffull);
  }
}

// weight [fi x fo] f32 -> transposed f16 [fo x kpad], zero-padded K
__global__ void wtrans_kernel(const float* __restrict__ w, int fi, int fo, int kpad,
                              _Float16* __restrict__ wt) {
  int t = blockIdx.x * blockDim.x + threadIdx.x;
  if (t >= fo * kpad) return;
  int o = t / kpad, c = t - o * kpad;
  wt[t] = (c < fi) ? (_Float16)w[c * fo + o] : (_Float16)0.0f;
}

__global__ void fill_kernel(float* __restrict__ p, int n, float v) {
  int i = blockIdx.x * blockDim.x + threadIdx.x;
  if (i < n) p[i] = v;
}

// fused EdgeConv: gather [xi | xj-xi] -> LDS, WMMA L1 + ReLU -> LDS, WMMA L2 + bias,
// atomic-max scatter into per-node output (max over K neighbors).
// Templated so tile counts are constexpr: A-fragment hoisted, accumulator array in registers.
template <int H, int K2P>
__global__ void edgeconv_kernel(const _Float16* __restrict__ xh, int ldx, int D,
                                const int* __restrict__ idx,
                                const _Float16* __restrict__ w1t, const float* __restrict__ b1,
                                const _Float16* __restrict__ w2t, const float* __restrict__ b2,
                                float* __restrict__ xout) {
  constexpr int NT = H / 16;
  __shared__ _Float16 s_et[4][16 * K2P];
  __shared__ _Float16 s_hid[4][16 * H];
  const int wid  = threadIdx.x >> 5;
  const int lane = threadIdx.x & 31;
  const int tile = blockIdx.x * 4 + wid;
  const int e0 = tile * 16;
  _Float16* et  = s_et[wid];
  _Float16* hid = s_hid[wid];
  // cooperative gather (wave-synchronous; LDS ops are in-order per wave)
  for (int t = lane; t < 16 * K2P; t += 32) {   // multiple of 32 -> no divergence
    int r = t / K2P, c = t - r * K2P;
    int e = e0 + r;
    int i = e / KNN;
    float v = 0.0f;
    if (c < D) v = (float)xh[i * ldx + c];
    else if (c < 2 * D) {
      int j = idx[e];
      int cc = c - D;
      v = (float)xh[j * ldx + cc] - (float)xh[i * ldx + cc];
    }
    et[r * K2P + c] = (_Float16)v;
  }
  __builtin_amdgcn_wave_barrier();
  const int hl = lane & 15, hi = lane >> 4;
  // layer 1: h = relu(E @ W1^T + b1) -> LDS
  {
    v8f acc[NT];
    v8f z = {};
#pragma unroll
    for (int nt = 0; nt < NT; ++nt) acc[nt] = z;
#pragma unroll
    for (int kb = 0; kb < K2P; kb += 32) {
      v16h a = load_a_frag(et, K2P, kb);
#pragma unroll
      for (int nt = 0; nt < NT; ++nt) {
        v16h b = load_bt_frag(w1t + (size_t)(nt * 16) * K2P, K2P, kb);
        acc[nt] = wmma16x16x32(a, b, acc[nt]);
      }
    }
#pragma unroll
    for (int nt = 0; nt < NT; ++nt) {
      int col = nt * 16 + hl;
      float bb = b1[col];
#pragma unroll
      for (int r = 0; r < 8; ++r) {
        float h = fmaxf(acc[nt][r] + bb, 0.0f);
        hid[(hi * 8 + r) * H + col] = (_Float16)h;
      }
    }
  }
  __builtin_amdgcn_wave_barrier();
  // layer 2: y = H @ W2^T + b2, then max-scatter into xout
  {
    v8f acc[NT];
    v8f z = {};
#pragma unroll
    for (int nt = 0; nt < NT; ++nt) acc[nt] = z;
#pragma unroll
    for (int kb = 0; kb < H; kb += 32) {
      v16h a = load_a_frag(hid, H, kb);
#pragma unroll
      for (int nt = 0; nt < NT; ++nt) {
        v16h b = load_bt_frag(w2t + (size_t)(nt * 16) * H, H, kb);
        acc[nt] = wmma16x16x32(a, b, acc[nt]);
      }
    }
#pragma unroll
    for (int nt = 0; nt < NT; ++nt) {
      int col = nt * 16 + hl;
      float bb = b2[col];
#pragma unroll
      for (int r = 0; r < 8; ++r) {
        int e = e0 + hi * 8 + r;
        int i = e / KNN;
        atomicMaxF(&xout[(size_t)i * H + col], acc[nt][r] + bb);
      }
    }
  }
}

// generic WMMA GEMM: out = (relu?)(A[n x kpad] @ Wt^T + bias), optional f16 copy.
// 16x64 block per wave, double-buffered fragment loads across k-steps.
template <int M>
__global__ void mlp_kernel(const _Float16* __restrict__ Ah, int kpad,
                           const _Float16* __restrict__ wt, const float* __restrict__ bias,
                           int relu, float* __restrict__ out, _Float16* __restrict__ outh) {
  constexpr int G = M / 64;
  const int wid  = threadIdx.x >> 5;
  const int tile = blockIdx.x * 4 + wid;
  const int rt = tile / G, cg = tile - rt * G;
  const _Float16* atile = Ah + (size_t)(rt * 16) * kpad;
  const _Float16* bt0   = wt + (size_t)(cg * 64) * kpad;
  const _Float16* bt1   = bt0 + (size_t)16 * kpad;
  const _Float16* bt2   = bt0 + (size_t)32 * kpad;
  const _Float16* bt3   = bt0 + (size_t)48 * kpad;
  v8f acc[4];
  {
    v8f z = {};
#pragma unroll
    for (int cb = 0; cb < 4; ++cb) acc[cb] = z;
  }
  v16h ca  = load_a_frag(atile, kpad, 0);
  v16h cb0 = load_bt_frag(bt0, kpad, 0);
  v16h cb1 = load_bt_frag(bt1, kpad, 0);
  v16h cb2 = load_bt_frag(bt2, kpad, 0);
  v16h cb3 = load_bt_frag(bt3, kpad, 0);
  for (int kb = 0; kb < kpad; kb += 32) {
    const int kn = (kb + 32 < kpad) ? kb + 32 : kb;
    v16h na  = load_a_frag(atile, kpad, kn);
    v16h nb0 = load_bt_frag(bt0, kpad, kn);
    v16h nb1 = load_bt_frag(bt1, kpad, kn);
    v16h nb2 = load_bt_frag(bt2, kpad, kn);
    v16h nb3 = load_bt_frag(bt3, kpad, kn);
    acc[0] = wmma16x16x32(ca, cb0, acc[0]);
    acc[1] = wmma16x16x32(ca, cb1, acc[1]);
    acc[2] = wmma16x16x32(ca, cb2, acc[2]);
    acc[3] = wmma16x16x32(ca, cb3, acc[3]);
    ca = na; cb0 = nb0; cb1 = nb1; cb2 = nb2; cb3 = nb3;
  }
  const int lane = threadIdx.x & 31;
  const int hl = lane & 15, hi = lane >> 4;
#pragma unroll
  for (int cb = 0; cb < 4; ++cb) {
    int col = cg * 64 + cb * 16 + hl;
    float bb = bias[col];
#pragma unroll
    for (int r = 0; r < 8; ++r) {
      int row = rt * 16 + hi * 8 + r;
      float v = acc[cb][r] + bb;
      if (relu) v = fmaxf(v, 0.0f);
      out[(size_t)row * M + col] = v;
      if (outh) outh[(size_t)row * M + col] = (_Float16)v;
    }
  }
}

__global__ void concat_kernel(const float* __restrict__ x1, const float* __restrict__ x2,
                              const float* __restrict__ x3, const float* __restrict__ x4,
                              _Float16* __restrict__ xcat, int n) {
  int t = blockIdx.x * blockDim.x + threadIdx.x;
  if (t >= n * 320) return;
  int i = t / 320, c = t - i * 320;
  float v;
  if (c < 64)       v = x1[i * 64 + c];
  else if (c < 128) v = x2[i * 64 + (c - 64)];
  else if (c < 192) v = x3[i * 64 + (c - 128)];
  else              v = x4[i * 128 + (c - 192)];
  xcat[t] = (_Float16)v;
}

__global__ void final3_kernel(const float* __restrict__ y, const float* __restrict__ w,
                              const float* __restrict__ b, float* __restrict__ out, int n) {
  int i = blockIdx.x * blockDim.x + threadIdx.x;
  if (i >= n) return;
  float a0 = b[0], a1 = b[1], a2 = b[2];
  for (int k = 0; k < 128; ++k) {
    float v = y[(size_t)i * 128 + k];
    a0 += v * w[k * 3 + 0];
    a1 += v * w[k * 3 + 1];
    a2 += v * w[k * 3 + 2];
  }
  out[i * 3 + 0] = a0; out[i * 3 + 1] = a1; out[i * 3 + 2] = a2;
}

__global__ void edges_kernel(const int* __restrict__ idx_all, int* __restrict__ eout) {
  int t = blockIdx.x * blockDim.x + threadIdx.x;
  if (t >= 4 * NEDGE) return;
  int l = t / NEDGE, e = t - l * NEDGE;
  int* base = eout + (size_t)l * 2 * NEDGE;
  base[e] = idx_all[(size_t)l * NEDGE + e];  // src
  base[NEDGE + e] = e / KNN;                 // dst
}

// ---------------- host orchestration ----------------

extern "C" void kernel_launch(void* const* d_in, const int* in_sizes, int n_in,
                              void* d_out, int out_size, void* d_ws, size_t ws_size,
                              hipStream_t stream) {
  (void)in_sizes; (void)n_in; (void)out_size;
  const int N = NPTS;
  const float* pos  = (const float*)d_in[0];
  const float* c1w1 = (const float*)d_in[2];  const float* c1b1 = (const float*)d_in[3];
  const float* c1w2 = (const float*)d_in[4];  const float* c1b2 = (const float*)d_in[5];
  const float* c2w1 = (const float*)d_in[6];  const float* c2b1 = (const float*)d_in[7];
  const float* c2w2 = (const float*)d_in[8];  const float* c2b2 = (const float*)d_in[9];
  const float* c3w1 = (const float*)d_in[10]; const float* c3b1 = (const float*)d_in[11];
  const float* c3w2 = (const float*)d_in[12]; const float* c3b2 = (const float*)d_in[13];
  const float* c4w1 = (const float*)d_in[14]; const float* c4b1 = (const float*)d_in[15];
  const float* c4w2 = (const float*)d_in[16]; const float* c4b2 = (const float*)d_in[17];
  const float* m1w  = (const float*)d_in[18]; const float* m1b  = (const float*)d_in[19];
  const float* m2w  = (const float*)d_in[20]; const float* m2b  = (const float*)d_in[21];
  const float* m3w  = (const float*)d_in[22]; const float* m3b  = (const float*)d_in[23];

  char* ws = (char*)d_ws;
  size_t off = 0;
  auto carve = [&](size_t bytes) {
    size_t o = off; off = (off + bytes + 255) & ~(size_t)255; return o;
  };
  float*    sq    = (float*)   (ws + carve((size_t)N * 4));
  _Float16* xh    = (_Float16*)(ws + carve((size_t)N * 128 * 2));
  int*      idxa  = (int*)     (ws + carve((size_t)4 * NEDGE * 4));
  float*    x1    = (float*)   (ws + carve((size_t)N * 64 * 4));
  float*    x2    = (float*)   (ws + carve((size_t)N * 64 * 4));
  float*    x3    = (float*)   (ws + carve((size_t)N * 64 * 4));
  float*    x4    = (float*)   (ws + carve((size_t)N * 128 * 4));
  _Float16* xcat  = (_Float16*)(ws + carve((size_t)N * 320 * 2));
  float*    y1    = (float*)   (ws + carve((size_t)N * 256 * 4));
  _Float16* y1h   = (_Float16*)(ws + carve((size_t)N * 256 * 2));
  float*    y2    = (float*)   (ws + carve((size_t)N * 128 * 4));
  _Float16* w1t   = (_Float16*)(ws + carve((size_t)128 * 256 * 2));
  _Float16* w2t   = (_Float16*)(ws + carve((size_t)128 * 128 * 2));
  _Float16* m1t   = (_Float16*)(ws + carve((size_t)256 * 320 * 2));
  _Float16* m2t   = (_Float16*)(ws + carve((size_t)128 * 256 * 2));
  float*    dbuf  = (float*)   (ws + off);

  size_t avail = (ws_size > off) ? (ws_size - off) : 0;
  int SR = (int)(avail / ((size_t)N * 4));
  SR &= ~31;                              // 2-row-tile (32-row) blocking in dist_kernel
  if (SR > 2048) SR = 2048;
  if (SR < 32) SR = 32;

  struct Layer {
    const float* xin; int D, Dpad;
    const float* w1; const float* b1; const float* w2; const float* b2;
    int H, k2pad; float* xout;
  };
  Layer L[4] = {
    { pos, 3, 32, c1w1, c1b1, c1w2, c1b2,  64,  32, x1 },
    { x1, 64, 64, c2w1, c2b1, c2w2, c2b2,  64, 128, x2 },
    { x2, 64, 64, c3w1, c3b1, c3w2, c3b2,  64, 128, x3 },
    { x3, 64, 64, c4w1, c4b1, c4w2, c4b2, 128, 128, x4 },
  };

  const int ecBlocks = (NEDGE / 16) / 4;  // 5120, exact

  for (int l = 0; l < 4; ++l) {
    const Layer& P = L[l];
    prep_kernel<<<(N + 255) / 256, 256, 0, stream>>>(P.xin, N, P.D, P.Dpad, sq, xh);
    int* idxL = idxa + (size_t)l * NEDGE;
    for (int s = 0; s < N; s += SR) {
      int rows = N - s; if (rows > SR) rows = SR;           // multiple of 32
      int tiles = (rows >> 5) * (N >> 6);                   // multiple of 4 (N>>6 = 256)
      dist_kernel<<<tiles / 4, 128, 0, stream>>>(xh, sq, N, P.Dpad, s, dbuf);
      topk_kernel<<<(rows + 3) / 4, 128, 0, stream>>>(dbuf, N, s, rows, idxL);
    }
    wtrans_kernel<<<(P.H * P.k2pad + 255) / 256, 256, 0, stream>>>(P.w1, 2 * P.D, P.H, P.k2pad, w1t);
    wtrans_kernel<<<(P.H * P.H + 255) / 256, 256, 0, stream>>>(P.w2, P.H, P.H, P.H, w2t);
    fill_kernel<<<(N * P.H + 255) / 256, 256, 0, stream>>>(P.xout, N * P.H, -__builtin_inff());
    if (P.H == 64 && P.k2pad == 32)
      edgeconv_kernel<64, 32><<<ecBlocks, 128, 0, stream>>>(xh, P.Dpad, P.D, idxL,
                                                            w1t, P.b1, w2t, P.b2, P.xout);
    else if (P.H == 64 && P.k2pad == 128)
      edgeconv_kernel<64, 128><<<ecBlocks, 128, 0, stream>>>(xh, P.Dpad, P.D, idxL,
                                                             w1t, P.b1, w2t, P.b2, P.xout);
    else
      edgeconv_kernel<128, 128><<<ecBlocks, 128, 0, stream>>>(xh, P.Dpad, P.D, idxL,
                                                              w1t, P.b1, w2t, P.b2, P.xout);
  }

  // head MLP
  concat_kernel<<<(N * 320 + 255) / 256, 256, 0, stream>>>(x1, x2, x3, x4, xcat, N);
  wtrans_kernel<<<(256 * 320 + 255) / 256, 256, 0, stream>>>(m1w, 320, 256, 320, m1t);
  mlp_kernel<256><<<((N / 16) * (256 / 64)) / 4, 128, 0, stream>>>(xcat, 320, m1t, m1b, 1, y1, y1h);
  wtrans_kernel<<<(128 * 256 + 255) / 256, 256, 0, stream>>>(m2w, 256, 128, 256, m2t);
  mlp_kernel<128><<<((N / 16) * (128 / 64)) / 4, 128, 0, stream>>>(y1h, 256, m2t, m2b, 1, y2,
                                                                   (_Float16*)nullptr);
  final3_kernel<<<(N + 255) / 256, 256, 0, stream>>>(y2, m3w, m3b, (float*)d_out, N);

  // edge index outputs (int32 written into the concatenated output buffer)
  int* eout = (int*)((float*)d_out + (size_t)N * 3);
  edges_kernel<<<(4 * NEDGE + 255) / 256, 256, 0, stream>>>(idxa, eout);
}
